// MultiHeadAttention_7301444403677
// MI455X (gfx1250) — matrix operand order repaired and checked by reading, not compiled, not run on hardware
//
#include <hip/hip_runtime.h>

#define B_   2
#define L_   2048
#define D_   1024
#define H_   16
#define HD_  64
#define M_   (B_ * L_)   // 4096 tokens

typedef __attribute__((ext_vector_type(16))) _Float16 v16h;
typedef __attribute__((ext_vector_type(8)))  _Float16 v8h;
typedef __attribute__((ext_vector_type(8)))  float    v8f;

__device__ __forceinline__ v8f zero_v8f() {
  v8f z;
#pragma unroll
  for (int i = 0; i < 8; ++i) z[i] = 0.0f;
  return z;
}

__device__ __forceinline__ v8h zero_v8h() {
  v8h z;
#pragma unroll
  for (int i = 0; i < 8; ++i) z[i] = (_Float16)0.0f;
  return z;
}

__device__ __forceinline__ v8f wmma_f16(v16h a, v16h b, v8f c) {
  // (neg_a, A, neg_b, B, c_mod, C, reuse_a, reuse_b)
  return __builtin_amdgcn_wmma_f32_16x16x32_f16(false, a, false, b, (short)0, c,
                                                false, false);
}

// A-fragment (16x32 f16): halves 0..7 = K[g*16..g*16+7], halves 8..15 = K[g*16+8..g*16+15]
__device__ __forceinline__ v16h frag_a_f16(const _Float16* p0, const _Float16* p1) {
  v8h a = *(const v8h*)p0;
  v8h b = *(const v8h*)p1;
  v16h r;
#pragma unroll
  for (int i = 0; i < 8; ++i) { r[i] = a[i]; r[i + 8] = b[i]; }
  return r;
}

// ---------------------------------------------------------------------------
// Kernel 1: weight f32 -> f16 transposed, LDS-tiled so both global accesses
// are coalesced.  WT[n][k] = W[k][n].
// ---------------------------------------------------------------------------
__global__ void __launch_bounds__(256)
wt_cvt_kernel(const float* __restrict__ W, _Float16* __restrict__ WT) {
  __shared__ _Float16 tile[32][33];
  const int tx = threadIdx.x & 31;
  const int ty = threadIdx.x >> 5;        // 0..7
  const int k0 = blockIdx.x * 32;         // source row block
  const int n0 = blockIdx.y * 32;         // source col block
#pragma unroll
  for (int i = 0; i < 4; ++i) {
    const int r = ty + i * 8;
    tile[r][tx] = (_Float16)W[(size_t)(k0 + r) * D_ + n0 + tx];   // coalesced read
  }
  __syncthreads();
#pragma unroll
  for (int i = 0; i < 4; ++i) {
    const int r = ty + i * 8;
    WT[(size_t)(n0 + r) * D_ + k0 + tx] = tile[tx][r];            // coalesced write
  }
}

// ---------------------------------------------------------------------------
// Kernel 2/5: GEMM  Out = A(4096x1024) * W + bias  (W given as WT, K-major)
//   A_F16   : 0 = A is f32 (convert while staging), 1 = A is f16
//   HEAD_OUT: 1 = write f16 head-major (B,H,L,HD) with scale, 0 = f32 flat
// Block = 64(M) x 256(N), 8 waves, each wave 32x64 (2x4 WMMA tiles).
// A panel (64x32) staged cooperatively in LDS each K-step: removes the 4x
// redundant per-wave A loads (4 waves share each M-group).
// ---------------------------------------------------------------------------
template <int A_F16, int HEAD_OUT>
__global__ void __launch_bounds__(256)
gemm_bias_kernel(const void* __restrict__ Aop, const _Float16* __restrict__ WT,
                 const float* __restrict__ bias, void* __restrict__ Out,
                 float scale) {
  __shared__ _Float16 sA[64][40];   // 64 rows x 32 k (pitch 80B)

  const int lane = threadIdx.x & 31;
  const int wid  = threadIdx.x >> 5;
  const int ln16 = lane & 15;
  const int g8   = (lane >> 4) * 8;
  const int g16  = (lane >> 4) * 16;

  const int mBase = blockIdx.y * 64;
  const int mloc  = (wid >> 2) * 32;               // wave M offset in block
  const int n0    = blockIdx.x * 256 + (wid & 3) * 64;

  const int ar = threadIdx.x >> 2;                 // staging row 0..63
  const int ac = (threadIdx.x & 3) * 8;            // staging col 0,8,16,24

  v8f acc[2][4];
#pragma unroll
  for (int i = 0; i < 2; ++i)
#pragma unroll
    for (int j = 0; j < 4; ++j) acc[i][j] = zero_v8f();

  for (int kk = 0; kk < D_; kk += 32) {
    // --- cooperative A-panel staging (convert f32->f16 if needed) ---
    if (A_F16) {
      const _Float16* Ah = (const _Float16*)Aop;
      *(v8h*)&sA[ar][ac] = *(const v8h*)(Ah + (size_t)(mBase + ar) * D_ + kk + ac);
    } else {
      const float* Af = (const float*)Aop + (size_t)(mBase + ar) * D_ + kk + ac;
      v8h t;
#pragma unroll
      for (int i = 0; i < 8; ++i) t[i] = (_Float16)Af[i];
      *(v8h*)&sA[ar][ac] = t;
    }
    __syncthreads();

    v16h af[2], bf[4];
#pragma unroll
    for (int mt = 0; mt < 2; ++mt) {
      const _Float16* arow = &sA[mloc + mt * 16 + ln16][0];
      af[mt] = frag_a_f16(arow + g16, arow + g16 + 8);
    }
#pragma unroll
    for (int nt = 0; nt < 4; ++nt) {
      const size_t wrow = (size_t)(n0 + nt * 16 + ln16);
      bf[nt] = *(const v16h*)(WT + wrow * D_ + kk + g16);
    }
#pragma unroll
    for (int mt = 0; mt < 2; ++mt)
#pragma unroll
      for (int nt = 0; nt < 4; ++nt)
        acc[mt][nt] = wmma_f16(af[mt], bf[nt], acc[mt][nt]);
    __syncthreads();
  }

  float bn[4];
#pragma unroll
  for (int nt = 0; nt < 4; ++nt) bn[nt] = bias[n0 + nt * 16 + ln16];

#pragma unroll
  for (int mt = 0; mt < 2; ++mt) {
#pragma unroll
    for (int nt = 0; nt < 4; ++nt) {
#pragma unroll
      for (int r = 0; r < 8; ++r) {
        const int rowm = mBase + mloc + mt * 16 + ((lane < 16) ? r : r + 8);
        const int col  = n0 + nt * 16 + ln16;
        const float v  = acc[mt][nt][r] + bn[nt];
        if (HEAD_OUT) {
          const int b = rowm >> 11;        // / L_
          const int l = rowm & (L_ - 1);
          const int h = col >> 6;          // / HD_
          const int d = col & (HD_ - 1);
          ((_Float16*)Out)[((size_t)(b * H_ + h) * L_ + l) * HD_ + d] =
              (_Float16)(v * scale);
        } else {
          ((float*)Out)[(size_t)rowm * D_ + col] = v;
        }
      }
    }
  }
}

// ---------------------------------------------------------------------------
// Kernel 4: flash attention, transposed formulation with double-buffered
// LDS staging (one barrier per key chunk; next chunk's global loads are
// issued before the current chunk's compute so latency hides under WMMA).
//   S^T = K x Q^T  (M=key, N=query)  -> per-lane softmax (query = lane%16)
//   O^T = V^T x P  (M=head-dim, N=query) -> per-lane scalar rescale
// Grid = (L/128, B*H), 8 waves x 16 query rows, keys in chunks of 32.
// ---------------------------------------------------------------------------
__global__ void __launch_bounds__(256)
flash_attn_kernel(const _Float16* __restrict__ Qh, const _Float16* __restrict__ Kh,
                  const _Float16* __restrict__ Vh, _Float16* __restrict__ AO) {
  __shared__ _Float16 sK[2][32][80];    // 32 keys x 64 dims (pitch 160B)
  __shared__ _Float16 sVT[2][64][48];   // V^T: 64 dims x 32 keys (pitch 96B)

  const int lane = threadIdx.x & 31;
  const int wid  = threadIdx.x >> 5;
  const int ln16 = lane & 15;
  const int g8   = (lane >> 4) * 8;
  const int g16  = (lane >> 4) * 16;

  const int bh = blockIdx.y;           // b*H + h
  const int b  = bh >> 4;
  const int h  = bh & (H_ - 1);
  const size_t headoff = (size_t)bh * L_ * HD_;
  const _Float16* Qp = Qh + headoff;
  const _Float16* Kp = Kh + headoff;
  const _Float16* Vp = Vh + headoff;

  const int q0 = blockIdx.x * 128 + wid * 16;

  // Q as B-fragment of S^T: qb[ks] halves = Q[q][ks*32 + g*16 + 0..15]
  v16h qb[2];
#pragma unroll
  for (int ks = 0; ks < 2; ++ks)
    qb[ks] = *(const v16h*)(Qp + (size_t)(q0 + ln16) * HD_ + ks * 32 + g16);

  v8f o[4];                            // O^T: 4 head-dim tiles x 16 queries
#pragma unroll
  for (int i = 0; i < 4; ++i) o[i] = zero_v8f();
  float mrun = -3.0e38f;
  float lrun = 0.0f;

  const int sr = threadIdx.x >> 3;        // staging row 0..31
  const int sc = (threadIdx.x & 7) * 8;   // staging col 0..56

  // --- prologue: stage chunk 0 into buffer 0 ---
  {
    v8h kv = *(const v8h*)(Kp + (size_t)sr * HD_ + sc);
    *(v8h*)&sK[0][sr][sc] = kv;
    v8h vv = *(const v8h*)(Vp + (size_t)sr * HD_ + sc);
#pragma unroll
    for (int i = 0; i < 8; ++i) sVT[0][sc + i][sr] = vv[i];
  }
  __syncthreads();

  const int NCH = L_ / 32;
  for (int ch = 0; ch < NCH; ++ch) {
    const int cur = ch & 1;
    const int nxt = cur ^ 1;
    const int kbn = (ch + 1) * 32;
    const bool have_next = (kbn < L_);

    // --- issue next chunk's global loads early (latency hides under WMMA) ---
    v8h kn = zero_v8h(), vn = zero_v8h();
    if (have_next) {
      kn = *(const v8h*)(Kp + (size_t)(kbn + sr) * HD_ + sc);
      vn = *(const v8h*)(Vp + (size_t)(kbn + sr) * HD_ + sc);
      if (kbn + 32 < L_) {
        __builtin_prefetch((const void*)(Kp + (size_t)(kbn + 32 + sr) * HD_ + sc), 0, 0);
        __builtin_prefetch((const void*)(Vp + (size_t)(kbn + 32 + sr) * HD_ + sc), 0, 0);
      }
    }

    // --- S^T tiles: key tile nt (M), queries (N) ---
    v8f s[2];
    s[0] = zero_v8f();
    s[1] = zero_v8f();
#pragma unroll
    for (int nt = 0; nt < 2; ++nt) {
      const _Float16* kr = &sK[cur][nt * 16 + ln16][0];
#pragma unroll
      for (int ks = 0; ks < 2; ++ks) {
        v16h ka = frag_a_f16(kr + ks * 32 + g16, kr + ks * 32 + g16 + 8);
        s[nt] = wmma_f16(ka, qb[ks], s[nt]);
      }
    }

    // --- online softmax, all per-lane (query = lane%16) ---
    float t = fmaxf(s[0][0], s[1][0]);
#pragma unroll
    for (int r = 1; r < 8; ++r) t = fmaxf(t, fmaxf(s[0][r], s[1][r]));
    t = fmaxf(t, __shfl_xor(t, 16, 32));       // other half holds the other 16 keys
    const float mnew  = fmaxf(mrun, t);
    const float alpha = __expf(mrun - mnew);
    mrun = mnew;

    float ps = 0.0f;
    v8h p0h, p1h;                               // local probs, f16
#pragma unroll
    for (int r = 0; r < 8; ++r) {
      const float p0 = __expf(s[0][r] - mnew);
      const float p1 = __expf(s[1][r] - mnew);
      p0h[r] = (_Float16)p0;
      p1h[r] = (_Float16)p1;
      ps += p0 + p1;
    }
    ps += __shfl_xor(ps, 16, 32);
    lrun = lrun * alpha + ps;
#pragma unroll
    for (int nt = 0; nt < 4; ++nt)
#pragma unroll
      for (int r = 0; r < 8; ++r) o[nt][r] *= alpha;

    // --- build P B-fragment: exchange packed f16 across half-waves ---
    union U { v8h h; int w[4]; };
    U l0, l1, x0, x1;
    l0.h = p0h;
    l1.h = p1h;
#pragma unroll
    for (int j = 0; j < 4; ++j) {
      x0.w[j] = __shfl_xor(l0.w[j], 16, 32);
      x1.w[j] = __shfl_xor(l1.w[j], 16, 32);
    }
    const bool lo = (lane < 16);
    v16h pb;   // halves 0..15 = P[key g*16 + 0..15][q]
#pragma unroll
    for (int i = 0; i < 8; ++i) {
      pb[i]     = lo ? l0.h[i] : x1.h[i];   // keys 0-7   | keys 16-23
      pb[8 + i] = lo ? x0.h[i] : l1.h[i];   // keys 8-15  | keys 24-31
    }

    // --- O^T += V^T x P : 4 head-dim tiles ---
#pragma unroll
    for (int nt = 0; nt < 4; ++nt) {
      const _Float16* vr = &sVT[cur][nt * 16 + ln16][0];
      v16h va = frag_a_f16(vr + g16, vr + g16 + 8);
      o[nt] = wmma_f16(va, pb, o[nt]);
    }

    // --- store next chunk into the other buffer, then single barrier ---
    if (have_next) {
      *(v8h*)&sK[nxt][sr][sc] = kn;
#pragma unroll
      for (int i = 0; i < 8; ++i) sVT[nxt][sc + i][sr] = vn[i];
    }
    __syncthreads();
  }

  // --- epilogue: normalize per lane, pack 8 consecutive dims, b128 stores ---
  const float inv = 1.0f / lrun;
  const int   q   = q0 + ln16;
  const int   db  = (lane < 16) ? 0 : 8;
  _Float16* aorow = AO + (size_t)(b * L_ + q) * D_ + h * HD_;
#pragma unroll
  for (int nt = 0; nt < 4; ++nt) {
    v8h oh;
#pragma unroll
    for (int r = 0; r < 8; ++r) oh[r] = (_Float16)(o[nt][r] * inv);
    *(v8h*)(aorow + nt * 16 + db) = oh;
  }
}

// ---------------------------------------------------------------------------
extern "C" void kernel_launch(void* const* d_in, const int* in_sizes, int n_in,
                              void* d_out, int out_size, void* d_ws, size_t ws_size,
                              hipStream_t stream) {
  (void)in_sizes; (void)n_in; (void)out_size; (void)ws_size;
  const float* q  = (const float*)d_in[0];
  const float* k  = (const float*)d_in[1];
  const float* v  = (const float*)d_in[2];
  const float* Wq = (const float*)d_in[3];
  const float* bq = (const float*)d_in[4];
  const float* Wk = (const float*)d_in[5];
  const float* bk = (const float*)d_in[6];
  const float* Wv = (const float*)d_in[7];
  const float* bv = (const float*)d_in[8];
  const float* Wo = (const float*)d_in[9];
  const float* bo = (const float*)d_in[10];

  char* ws = (char*)d_ws;
  const size_t WT_BYTES = (size_t)D_ * D_ * sizeof(_Float16);            // 2 MB
  const size_t HB_BYTES = (size_t)B_ * H_ * L_ * HD_ * sizeof(_Float16); // 8 MB
  _Float16* WqT = (_Float16*)(ws + 0 * WT_BYTES);
  _Float16* WkT = (_Float16*)(ws + 1 * WT_BYTES);
  _Float16* WvT = (_Float16*)(ws + 2 * WT_BYTES);
  _Float16* WoT = (_Float16*)(ws + 3 * WT_BYTES);
  _Float16* Qhp = (_Float16*)(ws + 4 * WT_BYTES);
  _Float16* Khp = (_Float16*)(ws + 4 * WT_BYTES + 1 * HB_BYTES);
  _Float16* Vhp = (_Float16*)(ws + 4 * WT_BYTES + 2 * HB_BYTES);
  _Float16* AO  = (_Float16*)(ws + 4 * WT_BYTES + 3 * HB_BYTES);

  const dim3 tgrid(D_ / 32, D_ / 32);
  wt_cvt_kernel<<<tgrid, 256, 0, stream>>>(Wq, WqT);
  wt_cvt_kernel<<<tgrid, 256, 0, stream>>>(Wk, WkT);
  wt_cvt_kernel<<<tgrid, 256, 0, stream>>>(Wv, WvT);
  wt_cvt_kernel<<<tgrid, 256, 0, stream>>>(Wo, WoT);

  const dim3 ggrid(D_ / 256, M_ / 64);
  const float SCALE = 0.125f;  // hd^-0.5, folded into Q at projection time
  gemm_bias_kernel<0, 1><<<ggrid, 256, 0, stream>>>(q, WqT, bq, Qhp, SCALE);
  gemm_bias_kernel<0, 1><<<ggrid, 256, 0, stream>>>(k, WkT, bk, Khp, 1.0f);
  gemm_bias_kernel<0, 1><<<ggrid, 256, 0, stream>>>(v, WvT, bv, Vhp, 1.0f);

  flash_attn_kernel<<<dim3(L_ / 128, B_ * H_), 256, 0, stream>>>(Qhp, Khp, Vhp, AO);

  gemm_bias_kernel<1, 0><<<ggrid, 256, 0, stream>>>(AO, WoT, bo, d_out, 1.0f);
}
